// RoleGatedSelfAttention_21088289423944
// MI455X (gfx1250) — matrix operand order repaired and checked by reading, not compiled, not run on hardware
//
#include <hip/hip_runtime.h>
#include <hip/hip_bf16.h>

// ---------------------------------------------------------------------------
// RoleGatedSelfAttention for MI455X (gfx1250, wave32, WMMA f32_16x16x32_f16)
//   B=2, L=2048, D=1024, H=16, hd=64
// Pipeline:
//   1) convert x, Wq/Wk/Wv/Wo fp32 -> f16 (workspace)
//   2) WMMA GEMM (LDS-free, operands loaded straight into WMMA reg layout):
//      Q/K/V = x @ W.T + b, stored f16 as [B,H,L,hd]
//   3) flash attention: 64x64 tiles, online softmax, role-delta + causal +
//      key-padding masks. K tiles double-buffered in LDS via CDNA5 async
//      global->LDS copies (ASYNCcnt / s_wait_asynccnt) when available;
//      V transposed once per tile in LDS; P transposed via wave-private LDS
//      strip; output ctx f16 [B,L,D]
//   4) WMMA GEMM: out = ctx @ Wo.T + bo (fp32)
// Workspace usage: 48 MB.
// ---------------------------------------------------------------------------

typedef __attribute__((ext_vector_type(16))) _Float16 v16h;
typedef __attribute__((ext_vector_type(8)))  _Float16 v8h;
typedef __attribute__((ext_vector_type(8)))  float    v8f;
typedef __attribute__((ext_vector_type(4)))  int      v4i;

#define D_MODEL   1024
#define NHEAD     16
#define HEAD_DIM  64
#define B_SZ      2
#define SEQ       2048
#define TOKENS    (B_SZ * SEQ)          // 4096
#define ATT_SCALE 0.125f                // 1/sqrt(64)
#define NEG_SENT  (-1.0e30f)

// CDNA5 async global->LDS copy (tracked by ASYNCcnt). Guarded: falls back to
// plain LDS stores if the toolchain does not expose the builtins.
#if defined(__has_builtin)
#if __has_builtin(__builtin_amdgcn_global_load_async_to_lds_b128)
#define HAVE_ASYNC_LDS 1
#endif
#endif
#ifndef HAVE_ASYNC_LDS
#define HAVE_ASYNC_LDS 0
#endif

#define AS1 __attribute__((address_space(1)))
#define AS3 __attribute__((address_space(3)))

__device__ __forceinline__ void wait_async_lds() {
#if HAVE_ASYNC_LDS
#if __has_builtin(__builtin_amdgcn_s_wait_asynccnt)
  __builtin_amdgcn_s_wait_asynccnt(0);
#else
  asm volatile("s_wait_asynccnt 0" ::: "memory");
#endif
#endif
}

union AOp { v16h v; v8h h[2]; };        // A operand: two 16B chunks per lane

// ---------------------------------------------------------------- fp32->f16
__global__ void cvt_f32_to_f16_kernel(const float* __restrict__ in,
                                      _Float16* __restrict__ out, int n) {
  int i = blockIdx.x * blockDim.x + threadIdx.x;
  if (i < n) out[i] = (_Float16)in[i];
}

// -------------------------------------------------------------- WMMA GEMM
// C[t, m] = sum_k A16[t,k] * W16[m,k] + bias[m]       (A row-major, W row-major)
// HEADS_OUT: store f16 into [B,H,L,hd]; else store fp32 into [T, D]
template <bool HEADS_OUT>
__global__ __launch_bounds__(128)
void gemm_xWT_kernel(const _Float16* __restrict__ A,
                     const _Float16* __restrict__ W,
                     const float* __restrict__ bias,
                     void* __restrict__ outp) {
  const int lane = threadIdx.x & 31;
  const int wave = threadIdx.x >> 5;
  const int lh   = lane >> 4;                 // half-wave (0/1)
  const int ln   = lane & 15;
  const int rowA = blockIdx.y * 64 + wave * 16 + ln;   // A row this lane loads
  const int colBase = blockIdx.x * 64;

  const v8f vzero = {0.f,0.f,0.f,0.f,0.f,0.f,0.f,0.f};
  v8f acc[4];
#pragma unroll
  for (int nt = 0; nt < 4; ++nt) acc[nt] = vzero;

  const _Float16* __restrict__ arow = A + (size_t)rowA * D_MODEL;

#pragma unroll 1
  for (int k0 = 0; k0 < D_MODEL; k0 += 32) {
    AOp a;                                    // A: 16B chunks @ k0+16*lh, k0+16*lh+8
    a.h[0] = *(const v8h*)(arow + k0 + 16 * lh);
    a.h[1] = *(const v8h*)(arow + k0 + 16 * lh + 8);
#pragma unroll
    for (int nt = 0; nt < 4; ++nt) {
      const _Float16* wrow = W + (size_t)(colBase + nt * 16 + ln) * D_MODEL;
      v16h bmat = *(const v16h*)(wrow + k0 + 16 * lh);  // B: 32B contiguous
      acc[nt] = __builtin_amdgcn_wmma_f32_16x16x32_f16(
          false, a.v, false, bmat, (short)0, acc[nt], false, false);
    }
  }

#pragma unroll
  for (int nt = 0; nt < 4; ++nt) {
    const int col = colBase + nt * 16 + ln;
    const float bv = bias[col];
#pragma unroll
    for (int v = 0; v < 8; ++v) {
      const int r = blockIdx.y * 64 + wave * 16 + v + 8 * lh;  // C/D row map
      const float val = acc[nt][v] + bv;
      if (HEADS_OUT) {
        const int bb = r >> 11, ll = r & 2047;
        const int hh = col >> 6, dd = col & 63;
        ((_Float16*)outp)[(((size_t)(bb * NHEAD + hh) * SEQ + ll) << 6) + dd] =
            (_Float16)val;
      } else {
        ((float*)outp)[(size_t)r * D_MODEL + col] = val;
      }
    }
  }
}

// --------------------------------------------------------- flash attention
__global__ __launch_bounds__(128)
void flash_attn_kernel(const _Float16* __restrict__ Q,
                       const _Float16* __restrict__ K,
                       const _Float16* __restrict__ V,
                       const int* __restrict__ role,
                       const unsigned char* __restrict__ kpad,
                       const float* __restrict__ deltap,
                       _Float16* __restrict__ ctx) {
  __shared__ _Float16 Klds[2][64 * 72];   // double-buffered K tile [k][d], pitch 72
  __shared__ _Float16 VT[64 * 72];        // V^T tile  [d][k], pitch 72 halves
  __shared__ _Float16 Pb[4 * 16 * 64];    // per-wave P strips [16][64]

  const int lane = threadIdx.x & 31;
  const int wave = threadIdx.x >> 5;
  const int lh = lane >> 4, ln = lane & 15;
  const int bh = blockIdx.y;              // b*NHEAD + h
  const int b  = bh >> 4;
  const int h  = bh & 15;
  const int q0 = blockIdx.x * 64;
  const float delta = *deltap;

  // Async (or fallback) staging of one 64x64 f16 K tile into Klds[buf].
  // 512 16-byte chunks, 4 per thread; per-lane LDS destinations give the
  // padded pitch-72 layout directly.
  auto preload_K = [&](int kb, int buf) {
    const _Float16* kbase = K + ((size_t)bh * SEQ + kb * 64) * HEAD_DIM;
#pragma unroll
    for (int i = 0; i < 4; ++i) {
      const int ch = threadIdx.x * 4 + i;
      const int r = ch >> 3, cc = ch & 7;           // row 0..63, chunk 0..7
      const _Float16* gsrc = kbase + r * 64 + cc * 8;
      _Float16* ldst = &Klds[buf][r * 72 + cc * 8];
#if HAVE_ASYNC_LDS
      __builtin_amdgcn_global_load_async_to_lds_b128(
          (AS1 v4i*)gsrc, (AS3 v4i*)ldst, 0, 0);
#else
      *(v8h*)ldst = *(const v8h*)gsrc;
#endif
    }
  };

  // Q operand (A layout) for the two K-substeps of hd=64
  const _Float16* qrow = Q + ((size_t)bh * SEQ + q0 + wave * 16 + ln) * HEAD_DIM;
  AOp aq[2];
#pragma unroll
  for (int s = 0; s < 2; ++s) {
    aq[s].h[0] = *(const v8h*)(qrow + 32 * s + 16 * lh);
    aq[s].h[1] = *(const v8h*)(qrow + 32 * s + 16 * lh + 8);
  }

  int roleQ[8];
#pragma unroll
  for (int v = 0; v < 8; ++v)
    roleQ[v] = role[b * SEQ + q0 + wave * 16 + v + 8 * lh];

  const v8f vzero = {0.f,0.f,0.f,0.f,0.f,0.f,0.f,0.f};
  v8f o[4];
  float m_[8], l_[8];
#pragma unroll
  for (int nt = 0; nt < 4; ++nt) o[nt] = vzero;
#pragma unroll
  for (int v = 0; v < 8; ++v) { m_[v] = NEG_SENT; l_[v] = 0.f; }

  _Float16* pstrip = Pb + wave * 16 * 64;
  const int nkb = blockIdx.x + 1;         // causal: key blocks 0..qblock

  preload_K(0, 0);                        // pipeline prologue

  for (int kb = 0; kb < nkb; ++kb) {
    const int cur = kb & 1;

    // ---- cooperative V -> VT (transposed) into LDS --------------------
    {
      const int r  = threadIdx.x >> 1;          // key row 0..63
      const int d0 = (threadIdx.x & 1) * 32;    // half of hd
      const _Float16* vrow =
          V + ((size_t)bh * SEQ + kb * 64 + r) * HEAD_DIM + d0;
      if (kb + 1 < nkb)
        __builtin_prefetch(V + ((size_t)bh * SEQ + (kb + 1) * 64 + r) * HEAD_DIM + d0, 0, 1);
      _Float16 tmp[32];
#pragma unroll
      for (int j = 0; j < 32; j += 8)
        *(v8h*)&tmp[j] = *(const v8h*)(vrow + j);
#pragma unroll
      for (int j = 0; j < 32; ++j)
        VT[(d0 + j) * 72 + r] = tmp[j];
    }

    wait_async_lds();                     // my K-tile async writes are done
    __syncthreads();                      // everyone's K writes + VT visible

    if (kb + 1 < nkb) preload_K(kb + 1, cur ^ 1);   // overlap DMA w/ compute

    // ---- per-key-column metadata -------------------------------------
    int kcol[4], roleK[4], padK[4];
#pragma unroll
    for (int nt = 0; nt < 4; ++nt) {
      kcol[nt]  = kb * 64 + nt * 16 + ln;
      roleK[nt] = role[b * SEQ + kcol[nt]];
      padK[nt]  = (int)kpad[b * SEQ + kcol[nt]];
    }

    // ---- S = Q @ K^T  (B operand from LDS K tile) ---------------------
    v8f s[4];
#pragma unroll
    for (int nt = 0; nt < 4; ++nt) s[nt] = vzero;
#pragma unroll
    for (int ss = 0; ss < 2; ++ss) {
#pragma unroll
      for (int nt = 0; nt < 4; ++nt) {
        v16h bk = *(const v16h*)(&Klds[cur][(nt * 16 + ln) * 72 + 32 * ss + 16 * lh]);
        s[nt] = __builtin_amdgcn_wmma_f32_16x16x32_f16(
            false, aq[ss].v, false, bk, (short)0, s[nt], false, false);
      }
    }

    // ---- scale + role delta + causal + padding masks ------------------
    float bm[8];
#pragma unroll
    for (int v = 0; v < 8; ++v) bm[v] = NEG_SENT;
#pragma unroll
    for (int nt = 0; nt < 4; ++nt) {
#pragma unroll
      for (int v = 0; v < 8; ++v) {
        float sv = s[nt][v] * ATT_SCALE + (roleQ[v] == roleK[nt] ? delta : 0.f);
        const int qi = q0 + wave * 16 + v + 8 * lh;
        if (padK[nt] || kcol[nt] > qi) sv = NEG_SENT;
        s[nt][v] = sv;
        bm[v] = fmaxf(bm[v], sv);
      }
    }
    // row max across the 16 lanes of each half-wave
#pragma unroll
    for (int v = 0; v < 8; ++v) {
#pragma unroll
      for (int off = 1; off < 16; off <<= 1)
        bm[v] = fmaxf(bm[v], __shfl_xor(bm[v], off, 32));
    }

    float alpha[8], nm[8];
#pragma unroll
    for (int v = 0; v < 8; ++v) {
      nm[v] = fmaxf(m_[v], bm[v]);
      alpha[v] = __expf(m_[v] - nm[v]);
      m_[v] = nm[v];
    }

    // ---- P = exp(S - m), rowsum, P -> LDS (C-layout -> strip) ---------
    float ps[8];
#pragma unroll
    for (int v = 0; v < 8; ++v) ps[v] = 0.f;
#pragma unroll
    for (int nt = 0; nt < 4; ++nt) {
#pragma unroll
      for (int v = 0; v < 8; ++v) {
        const float sv = s[nt][v];
        const float p = (sv <= -1.0e29f) ? 0.f : __expf(sv - nm[v]);
        ps[v] += p;
        pstrip[(v + 8 * lh) * 64 + nt * 16 + ln] = (_Float16)p;
      }
    }
#pragma unroll
    for (int v = 0; v < 8; ++v) {
#pragma unroll
      for (int off = 1; off < 16; off <<= 1)
        ps[v] += __shfl_xor(ps[v], off, 32);
      l_[v] = l_[v] * alpha[v] + ps[v];
    }
#pragma unroll
    for (int nt = 0; nt < 4; ++nt)
#pragma unroll
      for (int v = 0; v < 8; ++v) o[nt][v] *= alpha[v];

    // ---- O += P @ V ---------------------------------------------------
#pragma unroll
    for (int ss = 0; ss < 2; ++ss) {
      AOp ap;
      ap.h[0] = *(const v8h*)(pstrip + ln * 64 + 32 * ss + 16 * lh);
      ap.h[1] = *(const v8h*)(pstrip + ln * 64 + 32 * ss + 16 * lh + 8);
#pragma unroll
      for (int nt = 0; nt < 4; ++nt) {
        v16h bv = *(const v16h*)(&VT[(nt * 16 + ln) * 72 + 32 * ss + 16 * lh]);
        o[nt] = __builtin_amdgcn_wmma_f32_16x16x32_f16(
            false, ap.v, false, bv, (short)0, o[nt], false, false);
      }
    }
    __syncthreads();   // all waves done with VT / Klds[cur] before overwrite
  }

  // ---- epilogue: O / l  (fully-masked rows -> 0, matching nan_to_num) --
  float inv[8];
#pragma unroll
  for (int v = 0; v < 8; ++v) inv[v] = (l_[v] > 0.f) ? (1.f / l_[v]) : 0.f;
#pragma unroll
  for (int nt = 0; nt < 4; ++nt) {
#pragma unroll
    for (int v = 0; v < 8; ++v) {
      const int qi  = q0 + wave * 16 + v + 8 * lh;
      const int col = h * HEAD_DIM + nt * 16 + ln;
      ctx[(size_t)(b * SEQ + qi) * D_MODEL + col] = (_Float16)(o[nt][v] * inv[v]);
    }
  }
}

// ---------------------------------------------------------------- launcher
extern "C" void kernel_launch(void* const* d_in, const int* in_sizes, int n_in,
                              void* d_out, int out_size, void* d_ws, size_t ws_size,
                              hipStream_t stream) {
  (void)in_sizes; (void)n_in; (void)out_size; (void)ws_size;
  const float*         x     = (const float*)d_in[0];
  const int*           role  = (const int*)d_in[1];
  /* d_in[2] = causal attn_mask: triu(k=1), computed analytically in-kernel */
  const unsigned char* kpad  = (const unsigned char*)d_in[3];
  const float* Wq = (const float*)d_in[4];  const float* bq = (const float*)d_in[5];
  const float* Wk = (const float*)d_in[6];  const float* bk = (const float*)d_in[7];
  const float* Wv = (const float*)d_in[8];  const float* bv = (const float*)d_in[9];
  const float* Wo = (const float*)d_in[10]; const float* bo = (const float*)d_in[11];
  const float* delta = (const float*)d_in[12];

  char* ws = (char*)d_ws;                       // 48 MB layout
  _Float16* x16   = (_Float16*)(ws);                       //  0 .. 8 MB
  _Float16* Wq16  = (_Float16*)(ws + ( 8u << 20));
  _Float16* Wk16  = (_Float16*)(ws + (10u << 20));
  _Float16* Wv16  = (_Float16*)(ws + (12u << 20));
  _Float16* Wo16  = (_Float16*)(ws + (14u << 20));
  _Float16* Q16   = (_Float16*)(ws + (16u << 20));
  _Float16* K16   = (_Float16*)(ws + (24u << 20));
  _Float16* V16   = (_Float16*)(ws + (32u << 20));
  _Float16* ctx16 = (_Float16*)(ws + (40u << 20));

  const int nx = TOKENS * D_MODEL;              // 4,194,304
  const int nw = D_MODEL * D_MODEL;             // 1,048,576
  cvt_f32_to_f16_kernel<<<nx / 256, 256, 0, stream>>>(x,  x16,  nx);
  cvt_f32_to_f16_kernel<<<nw / 256, 256, 0, stream>>>(Wq, Wq16, nw);
  cvt_f32_to_f16_kernel<<<nw / 256, 256, 0, stream>>>(Wk, Wk16, nw);
  cvt_f32_to_f16_kernel<<<nw / 256, 256, 0, stream>>>(Wv, Wv16, nw);
  cvt_f32_to_f16_kernel<<<nw / 256, 256, 0, stream>>>(Wo, Wo16, nw);

  dim3 ggrid(D_MODEL / 64, TOKENS / 64);        // (16, 64)
  gemm_xWT_kernel<true><<<ggrid, 128, 0, stream>>>(x16, Wq16, bq, Q16);
  gemm_xWT_kernel<true><<<ggrid, 128, 0, stream>>>(x16, Wk16, bk, K16);
  gemm_xWT_kernel<true><<<ggrid, 128, 0, stream>>>(x16, Wv16, bv, V16);

  dim3 agrid(SEQ / 64, B_SZ * NHEAD);           // (32, 32)
  flash_attn_kernel<<<agrid, 128, 0, stream>>>(Q16, K16, V16, role, kpad,
                                               delta, ctx16);

  gemm_xWT_kernel<false><<<ggrid, 128, 0, stream>>>(ctx16, Wo16, bo, d_out);
}
